// StableContextualAttention_52467320488276
// MI455X (gfx1250) — compile-verified, matrix operand order checked by reading, not run on hardware
//
#include <hip/hip_runtime.h>
#include <hip/hip_bf16.h>

typedef __attribute__((ext_vector_type(16))) __bf16 v16bf;
typedef __attribute__((ext_vector_type(8)))  __bf16 v8bf;
typedef __attribute__((ext_vector_type(8)))  float  v8f;

#define BATCH 4
#define CH    256
#define NPIX  4096
#define EPSN  1e-12f
#define INV_TEMP 10.0f
#define KPAD  264              // 32-row K/X tile row pitch (bf16 elems), 528B: kills bank conflicts
#define VPAD  40               // 256-row V tile row pitch (bf16 elems), 80B

#define SHUF16(lo, hi) __builtin_shufflevector(lo, hi, 0,1,2,3,4,5,6,7,8,9,10,11,12,13,14,15)

// ---- CDNA5 async global->LDS copy (ASYNCcnt-tracked) -----------------------
__device__ __forceinline__ void async_ld16(uint32_t lds_off, const void* gaddr) {
    asm volatile("global_load_async_to_lds_b128 %0, %1, off"
                 :: "v"(lds_off), "v"((unsigned long long)(uintptr_t)gaddr)
                 : "memory");
}
__device__ __forceinline__ void wait_async0() {
    asm volatile("s_wait_asynccnt 0x0" ::: "memory");
}
__device__ __forceinline__ uint32_t lds_addr_of(const void* p) {
    // LDS flat aperture: addr[31:0] is the LDS byte offset (ISA 10.2)
    return (uint32_t)(uintptr_t)p;
}

// ---------------------------------------------------------------------------
// W f32 -> bf16 (one-time, 64K elements)
// ---------------------------------------------------------------------------
__global__ __launch_bounds__(256)
void wcvt_kernel(const float* __restrict__ Wc, __bf16* __restrict__ Wb) {
    const int i = blockIdx.x * 256 + threadIdx.x;
    Wb[i] = (__bf16)Wc[i];
}

// ---------------------------------------------------------------------------
// Projection as WMMA GEMM: one block = one batch x 32-pixel tile; 16 waves
// cover all 256 output channels. q = W@(x*m)+b, k = W@(x-x*m)+b, L2-normalized
// per pixel over channels. Outputs: Qn/Kn [B][N][C] bf16, V [B][C][N] bf16.
// ---------------------------------------------------------------------------
__global__ __launch_bounds__(512)
void proj_kernel(const float* __restrict__ x, const float* __restrict__ mask,
                 const __bf16* __restrict__ Wb, const float* __restrict__ bias,
                 __bf16* __restrict__ Qn, __bf16* __restrict__ Kn,
                 __bf16* __restrict__ Vb)
{
    __shared__ alignas(16) __bf16 XmT[32][KPAD];   // transposed masked input [p][c]
    __shared__ alignas(16) __bf16 XkT[32][KPAD];   // transposed known input  [p][c]
    __shared__ float sq[32], sk[32];

    const int blk = blockIdx.x;                    // 0 .. BATCH*NPIX/32-1
    const int bb  = blk / (NPIX / 32);
    const int n0  = (blk % (NPIX / 32)) * 32;
    const int tid = threadIdx.x;

    if (tid < 32) { sq[tid] = 0.f; sk[tid] = 0.f; }

    // stage inputs (transpose f32 [c][p] -> bf16 LDS [p][c]); also emit V
#pragma unroll
    for (int it = 0; it < 16; ++it) {
        const int q = tid + it * 512;              // 0..8191
        const int c = q >> 5, p = q & 31;
        const float mv = mask[bb * NPIX + n0 + p];
        const float xv = x[((size_t)(bb * CH + c)) * NPIX + n0 + p];
        const float xm = xv * mv;
        const float xk = xv - xm;
        XmT[p][c] = (__bf16)xm;
        XkT[p][c] = (__bf16)xk;
        Vb[((size_t)(bb * CH + c)) * NPIX + n0 + p] = (__bf16)xk;
    }
    __syncthreads();

    const int wave = tid >> 5, lane = tid & 31;
    const int lg = lane >> 4, ln = lane & 15;
    const int o0 = wave * 16;                      // this wave's output channels

    // A-fragments of W rows [o0..o0+15] x 32c, 8 K-steps
    v16bf wf[8];
#pragma unroll
    for (int ks = 0; ks < 8; ++ks) {
        const __bf16* wrow = Wb + (size_t)(o0 + ln) * CH + ks * 32;
        v8bf lo = *(const v8bf*)(wrow + 8 * lg);
        v8bf hi = *(const v8bf*)(wrow + 16 + 8 * lg);
        wf[ks] = SHUF16(lo, hi);
    }

    v8f q0 = {}, q1 = {}, k0 = {}, k1 = {};        // [16 o] x [16 p] tiles
#pragma unroll
    for (int ks = 0; ks < 8; ++ks) {
        v8bf a, b;
        a = *(const v8bf*)&XmT[ln][ks * 32 + 8 * lg];
        b = *(const v8bf*)&XmT[ln][ks * 32 + 16 + 8 * lg];
        v16bf bm0 = SHUF16(a, b);
        a = *(const v8bf*)&XmT[16 + ln][ks * 32 + 8 * lg];
        b = *(const v8bf*)&XmT[16 + ln][ks * 32 + 16 + 8 * lg];
        v16bf bm1 = SHUF16(a, b);
        a = *(const v8bf*)&XkT[ln][ks * 32 + 8 * lg];
        b = *(const v8bf*)&XkT[ln][ks * 32 + 16 + 8 * lg];
        v16bf bk0 = SHUF16(a, b);
        a = *(const v8bf*)&XkT[16 + ln][ks * 32 + 8 * lg];
        b = *(const v8bf*)&XkT[16 + ln][ks * 32 + 16 + 8 * lg];
        v16bf bk1 = SHUF16(a, b);
        q0 = __builtin_amdgcn_wmma_f32_16x16x32_bf16(false, wf[ks], false, bm0, (short)0, q0, false, false);
        q1 = __builtin_amdgcn_wmma_f32_16x16x32_bf16(false, wf[ks], false, bm1, (short)0, q1, false, false);
        k0 = __builtin_amdgcn_wmma_f32_16x16x32_bf16(false, wf[ks], false, bk0, (short)0, k0, false, false);
        k1 = __builtin_amdgcn_wmma_f32_16x16x32_bf16(false, wf[ks], false, bk1, (short)0, k1, false, false);
    }

    // bias, then per-pixel sum of squares (over all 256 channels via LDS atomics)
    float bv[8];
#pragma unroll
    for (int r = 0; r < 8; ++r) bv[r] = bias[o0 + r + 8 * lg];
    float pq0 = 0.f, pq1 = 0.f, pk0 = 0.f, pk1 = 0.f;
#pragma unroll
    for (int r = 0; r < 8; ++r) {
        q0[r] += bv[r]; q1[r] += bv[r]; k0[r] += bv[r]; k1[r] += bv[r];
        pq0 += q0[r] * q0[r]; pq1 += q1[r] * q1[r];
        pk0 += k0[r] * k0[r]; pk1 += k1[r] * k1[r];
    }
    atomicAdd(&sq[ln], pq0);      atomicAdd(&sq[16 + ln], pq1);
    atomicAdd(&sk[ln], pk0);      atomicAdd(&sk[16 + ln], pk1);
    __syncthreads();

    const float rq0 = 1.f / fmaxf(sqrtf(sq[ln]),      EPSN);
    const float rq1 = 1.f / fmaxf(sqrtf(sq[16 + ln]), EPSN);
    const float rk0 = 1.f / fmaxf(sqrtf(sk[ln]),      EPSN);
    const float rk1 = 1.f / fmaxf(sqrtf(sk[16 + ln]), EPSN);
#pragma unroll
    for (int r = 0; r < 8; ++r) {
        const int o = o0 + r + 8 * lg;
        Qn[((size_t)(bb * NPIX + n0 + ln)) * CH + o]      = (__bf16)(q0[r] * rq0);
        Qn[((size_t)(bb * NPIX + n0 + 16 + ln)) * CH + o] = (__bf16)(q1[r] * rq1);
        Kn[((size_t)(bb * NPIX + n0 + ln)) * CH + o]      = (__bf16)(k0[r] * rk0);
        Kn[((size_t)(bb * NPIX + n0 + 16 + ln)) * CH + o] = (__bf16)(k1[r] * rk1);
    }
}

// ---------------------------------------------------------------------------
// Flash attention: block = 4 waves x 16 query rows. K/V j-tiles staged in LDS
// by async global->LDS copies, double-buffered (prefetch t+1 under tile t's
// WMMAs, s_wait_asynccnt at iteration boundary).
// ---------------------------------------------------------------------------
__global__ __launch_bounds__(128)
void attn_kernel(const __bf16* __restrict__ Qn, const __bf16* __restrict__ Kn,
                 const __bf16* __restrict__ Vb, const float* __restrict__ x,
                 const float* __restrict__ mask, float* __restrict__ out)
{
    __shared__ alignas(16) __bf16 KtB[2][32][KPAD];    // [j][c] tiles
    __shared__ alignas(16) __bf16 VtB[2][256][VPAD];   // [c][j] tiles
    __shared__ alignas(16) __bf16 Pb[4][16][32];       // per-wave P transpose

    const int tid  = threadIdx.x;
    const int wave = tid >> 5;
    const int lane = tid & 31;
    const int lg   = lane >> 4;
    const int ln   = lane & 15;

    const int gw = blockIdx.x * 4 + wave;
    const int bb = gw / (NPIX / 16);
    const int i0 = (gw % (NPIX / 16)) * 16;

    const uint32_t ldsK0 = lds_addr_of(&KtB[0][0][0]);
    const uint32_t ldsV0 = lds_addr_of(&VtB[0][0][0]);
    const uint32_t kbufsz = 32 * KPAD * 2;
    const uint32_t vbufsz = 256 * VPAD * 2;

    // ---- Q fragments (A: row m=ln; elem e -> k=(e<8?e:e+8)+8*lg) ----------
    v16bf qf[8];
#pragma unroll
    for (int ks = 0; ks < 8; ++ks) {
        const __bf16* qrow = Qn + ((size_t)(bb * NPIX + i0 + ln)) * CH + ks * 32;
        v8bf lo = *(const v8bf*)(qrow + 8 * lg);
        v8bf hi = *(const v8bf*)(qrow + 16 + 8 * lg);
        qf[ks] = SHUF16(lo, hi);
    }

    v8f o[16] = {};
    float mrow[8], lrow[8];
#pragma unroll
    for (int r = 0; r < 8; ++r) { mrow[r] = -1e30f; lrow[r] = 0.f; }

    // ---- async tile loader: K tile 1024x16B chunks, V tile 1024x16B chunks
    auto load_tiles = [&](int jn, uint32_t ldsK, uint32_t ldsV) {
#pragma unroll
        for (int it = 0; it < 8; ++it) {
            const int q = tid + it * 128;
            const int row = q >> 5, ch = q & 31;        // K: 32 rows x 32 chunks
            async_ld16(ldsK + row * (KPAD * 2) + ch * 16,
                       Kn + ((size_t)(bb * NPIX + jn + row)) * CH + ch * 8);
        }
#pragma unroll
        for (int it = 0; it < 8; ++it) {
            const int q = tid + it * 128;
            const int row = q >> 2, ch = q & 3;         // V: 256 rows x 4 chunks
            async_ld16(ldsV + row * (VPAD * 2) + ch * 16,
                       Vb + ((size_t)(bb * CH + row)) * NPIX + jn + ch * 8);
        }
    };

    load_tiles(0, ldsK0, ldsV0);
    wait_async0();
    __syncthreads();

    for (int j0 = 0; j0 < NPIX; j0 += 32) {
        const int buf = (j0 >> 5) & 1;
        if (j0 + 32 < NPIX)                             // prefetch next tile
            load_tiles(j0 + 32, ldsK0 + (buf ^ 1) * kbufsz, ldsV0 + (buf ^ 1) * vbufsz);

        // ---- S = Q.K^T (16 x 32), K fragments from LDS --------------------
        v8f s0 = {}, s1 = {};
#pragma unroll
        for (int ks = 0; ks < 8; ++ks) {
            v8bf a, b;
            a = *(const v8bf*)&KtB[buf][ln][ks * 32 + 8 * lg];
            b = *(const v8bf*)&KtB[buf][ln][ks * 32 + 16 + 8 * lg];
            v16bf bf0 = SHUF16(a, b);
            a = *(const v8bf*)&KtB[buf][16 + ln][ks * 32 + 8 * lg];
            b = *(const v8bf*)&KtB[buf][16 + ln][ks * 32 + 16 + 8 * lg];
            v16bf bf1 = SHUF16(a, b);
            s0 = __builtin_amdgcn_wmma_f32_16x16x32_bf16(false, qf[ks], false, bf0, (short)0, s0, false, false);
            s1 = __builtin_amdgcn_wmma_f32_16x16x32_bf16(false, qf[ks], false, bf1, (short)0, s1, false, false);
        }

        // ---- online softmax ----------------------------------------------
        float alpha[8];
#pragma unroll
        for (int r = 0; r < 8; ++r) {
            const float a  = s0[r] * INV_TEMP;
            const float bs = s1[r] * INV_TEMP;
            float v = fmaxf(a, bs);
            v = fmaxf(v, __shfl_xor(v, 1, 16));
            v = fmaxf(v, __shfl_xor(v, 2, 16));
            v = fmaxf(v, __shfl_xor(v, 4, 16));
            v = fmaxf(v, __shfl_xor(v, 8, 16));
            const float mn = fmaxf(mrow[r], v);
            alpha[r] = __expf(mrow[r] - mn);
            const float p0 = __expf(a - mn);
            const float p1 = __expf(bs - mn);
            float ps = p0 + p1;
            ps += __shfl_xor(ps, 1, 16);
            ps += __shfl_xor(ps, 2, 16);
            ps += __shfl_xor(ps, 4, 16);
            ps += __shfl_xor(ps, 8, 16);
            lrow[r] = lrow[r] * alpha[r] + ps;
            mrow[r] = mn;
            Pb[wave][r + 8 * lg][ln]      = (__bf16)p0;
            Pb[wave][r + 8 * lg][16 + ln] = (__bf16)p1;
        }
#pragma unroll
        for (int f = 0; f < 16; ++f) {
#pragma unroll
            for (int r = 0; r < 8; ++r) o[f][r] *= alpha[r];
        }

        // ---- P: D-layout -> A-layout via LDS ------------------------------
        __syncthreads();
        v8bf plo = *(const v8bf*)&Pb[wave][ln][8 * lg];
        v8bf phi = *(const v8bf*)&Pb[wave][ln][16 + 8 * lg];
        v16bf pf = SHUF16(plo, phi);

        // ---- O += P @ V (V fragments from LDS) ----------------------------
#pragma unroll
        for (int f = 0; f < 16; ++f) {
            v8bf vlo = *(const v8bf*)&VtB[buf][f * 16 + ln][8 * lg];
            v8bf vhi = *(const v8bf*)&VtB[buf][f * 16 + ln][16 + 8 * lg];
            v16bf vf = SHUF16(vlo, vhi);
            o[f] = __builtin_amdgcn_wmma_f32_16x16x32_bf16(false, pf, false, vf, (short)0, o[f], false, false);
        }

        wait_async0();                                  // next tile landed
        __syncthreads();                                // all waves done with bufs
    }

    // ---- epilogue: out = x_known + (O / l) * mask --------------------------
    float invl[8];
#pragma unroll
    for (int r = 0; r < 8; ++r) invl[r] = 1.f / lrow[r];
#pragma unroll
    for (int f = 0; f < 16; ++f) {
        const int c = f * 16 + ln;
#pragma unroll
        for (int r = 0; r < 8; ++r) {
            const int i = i0 + r + 8 * lg;
            const size_t idx = ((size_t)(bb * CH + c)) * NPIX + i;
            const float mk = mask[bb * NPIX + i];
            const float xv = x[idx];
            out[idx] = xv * (1.f - mk) + o[f][r] * invl[r] * mk;
        }
    }
}

// ---------------------------------------------------------------------------
extern "C" void kernel_launch(void* const* d_in, const int* in_sizes, int n_in,
                              void* d_out, int out_size, void* d_ws, size_t ws_size,
                              hipStream_t stream) {
    const float* x    = (const float*)d_in[0];
    const float* mask = (const float*)d_in[1];
    const float* Wc   = (const float*)d_in[2];
    const float* bias = (const float*)d_in[3];
    float* out = (float*)d_out;

    __bf16* Qn = (__bf16*)d_ws;                          // 8 MB
    __bf16* Kn = Qn + (size_t)BATCH * NPIX * CH;         // 8 MB
    __bf16* Vb = Kn + (size_t)BATCH * NPIX * CH;         // 8 MB
    __bf16* Wb = Vb + (size_t)BATCH * NPIX * CH;         // 128 KB

    wcvt_kernel<<<CH * CH / 256, 256, 0, stream>>>(Wc, Wb);
    proj_kernel<<<BATCH * NPIX / 32, 512, 0, stream>>>(x, mask, Wb, bias, Qn, Kn, Vb);
    attn_kernel<<<(BATCH * NPIX / 16) / 4, 128, 0, stream>>>(Qn, Kn, Vb, x, mask, out);
}
